// MambaFPNHead_10857677324814
// MI455X (gfx1250) — compile-verified, matrix operand order checked by reading, not run on hardware
//
#include <hip/hip_runtime.h>
#include <hip/hip_bf16.h>
#include <math.h>

// ============================================================================
// MambaFPNHead for MI455X (gfx1250). All matmuls go through one WMMA GEMM
// kernel (v_wmma_f32_16x16x32_f16, f32 accumulate) with fused epilogues
// (bias / softplus / BN+ReLU). Fast path: unguarded float4 (b128) fragment
// loads, 2 WMMA per K-step (A reuse). Edge path: clamp+select (branchless).
// The selective scan is a per-(b,channel) register-resident loop.
// ============================================================================

typedef __attribute__((ext_vector_type(16))) _Float16 v16h;
typedef __attribute__((ext_vector_type(8)))  float    v8f;

static __device__ __forceinline__ float siluf(float x) {
  return x / (1.f + __expf(-x));
}

// Pack 16 contiguous-ish floats (two 8-float runs) into a f16 fragment.
static __device__ __forceinline__ v16h pack16(float4 x0, float4 x1,
                                              float4 x2, float4 x3) {
  v16h r;
  r[0]  = (_Float16)x0.x; r[1]  = (_Float16)x0.y;
  r[2]  = (_Float16)x0.z; r[3]  = (_Float16)x0.w;
  r[4]  = (_Float16)x1.x; r[5]  = (_Float16)x1.y;
  r[6]  = (_Float16)x1.z; r[7]  = (_Float16)x1.w;
  r[8]  = (_Float16)x2.x; r[9]  = (_Float16)x2.y;
  r[10] = (_Float16)x2.z; r[11] = (_Float16)x2.w;
  r[12] = (_Float16)x3.x; r[13] = (_Float16)x3.y;
  r[14] = (_Float16)x3.z; r[15] = (_Float16)x3.w;
  return r;
}

// ---------------------------------------------------------------------------
// WMMA GEMM: C[M,N] = A[M,K] (row stride lda) * W[N,K]^T (row stride ldw)
//   epi 0: + bias[n] (if bias != null)
//   epi 1: softplus(acc + bias[n])
//   epi 2: relu(acc * scale[n>>bnShift] + shift[n>>bnShift])
// One wave computes a 16x32 tile (two 16x16 N-tiles, one A fragment reused).
// Block = 4 waves along N. Fragment layouts per CDNA5 ISA 7.12.2:
// lane l holds row l%16; K(i) = (i>>3)*16 + (l>>4)*8 + (i&7).
// ---------------------------------------------------------------------------
__global__ void k_gemm(const float* __restrict__ A, int lda,
                       const float* __restrict__ Wt, int ldw,
                       const float* __restrict__ bias,
                       const float* __restrict__ scale,
                       const float* __restrict__ shift, int bnShift,
                       float* __restrict__ C, int ldc,
                       int M, int N, int K, int epi)
{
  int wave = threadIdx.x >> 5;
  int lane = threadIdx.x & 31;
  int tm  = blockIdx.x;
  int tn0 = (blockIdx.y * 4 + wave) * 2;     // first of two N-tiles
  if (tn0 * 16 >= N) return;                 // wave-uniform: EXEC all-ones below
  int row = lane & 15;
  int hh  = lane >> 4;
  int m  = tm * 16 + row;                    // A row loaded by this lane
  int n0 = tn0 * 16 + row;                   // W rows (= B columns)
  int n1 = n0 + 16;

  v8f acc0 = {0.f, 0.f, 0.f, 0.f, 0.f, 0.f, 0.f, 0.f};
  v8f acc1 = {0.f, 0.f, 0.f, 0.f, 0.f, 0.f, 0.f, 0.f};

  bool fast = ((tm + 1) * 16 <= M) && ((tn0 + 2) * 16 <= N) && ((K & 31) == 0);
  if (fast) {
    const float* ap  = A  + (size_t)m  * lda + hh * 8;
    const float* wp0 = Wt + (size_t)n0 * ldw + hh * 8;
    const float* wp1 = Wt + (size_t)n1 * ldw + hh * 8;
    for (int k0 = 0; k0 < K; k0 += 32, ap += 32, wp0 += 32, wp1 += 32) {
      const float4* pa = (const float4*)ap;
      const float4* pb0 = (const float4*)wp0;
      const float4* pb1 = (const float4*)wp1;
      v16h a  = pack16(pa[0],  pa[1],  pa[4],  pa[5]);   // +0,+4,+16,+20 floats
      v16h b0 = pack16(pb0[0], pb0[1], pb0[4], pb0[5]);
      v16h b1 = pack16(pb1[0], pb1[1], pb1[4], pb1[5]);
      acc0 = __builtin_amdgcn_wmma_f32_16x16x32_f16(
          false, a, false, b0, (short)0, acc0, false, false);
      acc1 = __builtin_amdgcn_wmma_f32_16x16x32_f16(
          false, a, false, b1, (short)0, acc1, false, false);
    }
  } else {
    // Branchless edge path: clamped loads + select-to-zero (no EXEC juggling).
    int mc  = m  < M ? m  : M - 1;
    int n0c = n0 < N ? n0 : N - 1;
    int n1c = n1 < N ? n1 : N - 1;
    bool mok  = (m  < M);
    bool n0ok = (n0 < N);
    bool n1ok = (n1 < N);
    const float* arow = A  + (size_t)mc  * lda;
    const float* wr0  = Wt + (size_t)n0c * ldw;
    const float* wr1  = Wt + (size_t)n1c * ldw;
    for (int k0 = 0; k0 < K; k0 += 32) {
      v16h a, b0, b1;
#pragma unroll
      for (int i = 0; i < 16; ++i) {
        int kk = k0 + ((i >> 3) << 4) + (hh << 3) + (i & 7);
        int kc = kk < K ? kk : K - 1;
        bool kok = (kk < K);
        float av  = arow[kc]; av  = (mok  && kok) ? av  : 0.f;
        float bv0 = wr0[kc];  bv0 = (n0ok && kok) ? bv0 : 0.f;
        float bv1 = wr1[kc];  bv1 = (n1ok && kok) ? bv1 : 0.f;
        a[i]  = (_Float16)av;
        b0[i] = (_Float16)bv0;
        b1[i] = (_Float16)bv1;
      }
      acc0 = __builtin_amdgcn_wmma_f32_16x16x32_f16(
          false, a, false, b0, (short)0, acc0, false, false);
      acc1 = __builtin_amdgcn_wmma_f32_16x16x32_f16(
          false, a, false, b1, (short)0, acc1, false, false);
    }
  }

  // C/D layout: VGPR r -> m = r + 8*hh, n = row (per tile).
  bool hasb = (bias != nullptr);
#pragma unroll
  for (int t = 0; t < 2; ++t) {
    v8f* accp = t ? &acc1 : &acc0;
    int nn = (tn0 + t) * 16 + row;
    if (nn >= N) continue;
    float sc = 0.f, sh = 0.f, bb = 0.f;
    if (epi == 2) { int c = nn >> bnShift; sc = scale[c]; sh = shift[c]; }
    else if (hasb || epi == 1) { bb = bias[nn]; }
#pragma unroll
    for (int r = 0; r < 8; ++r) {
      int mm = tm * 16 + r + hh * 8;
      if (mm < M) {
        float v = (*accp)[r];
        if (epi == 0)      { v += bb; }
        else if (epi == 1) { v += bb; v = (v > 20.f) ? v : log1pf(__expf(v)); }
        else               { v = v * sc + sh; v = v > 0.f ? v : 0.f; }
        C[(size_t)mm * ldc + nn] = v;
      }
    }
  }
}

// (B, C, L) -> (B, L, C)
__global__ void k_t_bcl_blc(const float* __restrict__ in, float* __restrict__ out,
                            int Bb, int Cc, int L)
{
  size_t tid = (size_t)blockIdx.x * blockDim.x + threadIdx.x;
  size_t total = (size_t)Bb * Cc * L;
  if (tid >= total) return;
  int c = (int)(tid % Cc);
  size_t t2 = tid / Cc;
  int l = (int)(t2 % L);
  int b = (int)(t2 / L);
  out[tid] = in[((size_t)b * Cc + c) * L + l];
}

// out[b,c,l] = (res ? res[b,c,l] : 0) + in[(b*L+l)*C + c]
__global__ void k_t_blc_bcl_add(const float* __restrict__ in, const float* __restrict__ res,
                                float* __restrict__ out, int Bb, int Cc, int L)
{
  size_t tid = (size_t)blockIdx.x * blockDim.x + threadIdx.x;
  size_t total = (size_t)Bb * Cc * L;
  if (tid >= total) return;
  int l = (int)(tid % L);
  size_t t2 = tid / L;
  int c = (int)(t2 % Cc);
  int b = (int)(t2 / Cc);
  float v = in[((size_t)b * L + l) * Cc + c];
  if (res) v += res[tid];
  out[tid] = v;
}

// Depthwise causal conv (D_CONV=4) + SiLU. xh lives inside xz with row
// stride ld (columns [0, di)). Output xc is contiguous [B, L, di].
__global__ void k_dwconv_silu(const float* __restrict__ xz, int ld,
                              const float* __restrict__ cw, const float* __restrict__ cb,
                              float* __restrict__ xc, int Bb, int L, int di)
{
  size_t tid = (size_t)blockIdx.x * blockDim.x + threadIdx.x;
  size_t total = (size_t)Bb * L * di;
  if (tid >= total) return;
  int d = (int)(tid % di);
  size_t t2 = tid / di;
  int t = (int)(t2 % L);
  int b = (int)(t2 / L);
  float acc = cb[d];
#pragma unroll
  for (int k = 0; k < 4; ++k) {
    int tt = t - 3 + k;
    if (tt >= 0) acc += xz[((size_t)b * L + tt) * ld + d] * cw[d * 4 + k];
  }
  xc[tid] = siluf(acc);
}

// Selective scan, one thread per (b, d) channel; h[16] in registers.
// Writes the fully gated output: y = (scan_y + x*Dp) * silu(z).
__global__ void k_scan(const float* __restrict__ dt, const float* __restrict__ xc,
                       const float* __restrict__ dbc, int ldd, int offB, int offC,
                       const float* __restrict__ xz, int ldz, int offz,
                       const float* __restrict__ A_log, const float* __restrict__ Dp,
                       float* __restrict__ y, int Bb, int L, int di)
{
  int idx = blockIdx.x * blockDim.x + threadIdx.x;
  if (idx >= Bb * di) return;
  int b = idx / di;
  int d = idx % di;
  float Av[16], h[16];
#pragma unroll
  for (int s = 0; s < 16; ++s) { Av[s] = -__expf(A_log[d * 16 + s]); h[s] = 0.f; }
  float Dd = Dp[d];
  for (int t = 0; t < L; ++t) {
    size_t rt = (size_t)b * L + t;
    float dtv = dt[rt * di + d];
    float xv  = xc[rt * di + d];
    const float* rowp = dbc + rt * ldd;
    float zv = xz[rt * ldz + offz + d];
    if (t + 1 < L) {  // global_prefetch_b8 for the next timestep's rows
      __builtin_prefetch((const void*)(dt + (rt + 1) * di + d), 0, 1);
      __builtin_prefetch((const void*)(xc + (rt + 1) * di + d), 0, 1);
    }
    float dtx = dtv * xv;
    float yv = 0.f;
#pragma unroll
    for (int s = 0; s < 16; ++s) {
      float dA = __expf(dtv * Av[s]);
      h[s] = h[s] * dA + dtx * rowp[offB + s];
      yv += h[s] * rowp[offC + s];
    }
    y[rt * di + d] = (yv + xv * Dd) * siluf(zv);
  }
}

// Bilinear resize (align-corners style, matching jnp.linspace mapping),
// writing into a channel slice [coff, coff+Cc) of a (B, outC, OH, OW) tensor.
__global__ void k_resize(const float* __restrict__ in, int Bb, int Cc, int H, int W,
                         float* __restrict__ out, int outC, int coff, int OH, int OW)
{
  size_t tid = (size_t)blockIdx.x * blockDim.x + threadIdx.x;
  size_t total = (size_t)Bb * Cc * OH * OW;
  if (tid >= total) return;
  int ox = (int)(tid % OW);
  size_t t2 = tid / OW;
  int oy = (int)(t2 % OH);
  size_t t3 = t2 / OH;
  int c = (int)(t3 % Cc);
  int b = (int)(t3 / Cc);
  float fy = (OH > 1) ? oy * (float)(H - 1) / (float)(OH - 1) : 0.f;
  float fx = (OW > 1) ? ox * (float)(W - 1) / (float)(OW - 1) : 0.f;
  int y0 = (int)floorf(fy); y0 = y0 < 0 ? 0 : (y0 > H - 1 ? H - 1 : y0);
  int x0 = (int)floorf(fx); x0 = x0 < 0 ? 0 : (x0 > W - 1 ? W - 1 : x0);
  int y1 = (y0 + 1 < H) ? y0 + 1 : H - 1;
  int x1 = (x0 + 1 < W) ? x0 + 1 : W - 1;
  float wy = fy - (float)y0;
  float wx = fx - (float)x0;
  const float* pc = in + ((size_t)b * Cc + c) * H * W;
  float tl = pc[y0 * W + x0], tr = pc[y0 * W + x1];
  float bl = pc[y1 * W + x0], br = pc[y1 * W + x1];
  float top = tl * (1.f - wx) + tr * wx;
  float bot = bl * (1.f - wx) + br * wx;
  out[((size_t)b * outC + coff + c) * OH * OW + (size_t)oy * OW + ox] =
      top * (1.f - wy) + bot * wy;
}

// ConvT 2x2 scatter: res is the GEMM result [B*H*W, O*4] with column
// n = o*4 + i*2 + j; out is (B, O, 2H, 2W).
__global__ void k_scatter_convT(const float* __restrict__ res, float* __restrict__ out,
                                int Bb, int O, int H, int W)
{
  size_t tid = (size_t)blockIdx.x * blockDim.x + threadIdx.x;
  size_t total = (size_t)Bb * O * 4 * H * W;
  if (tid >= total) return;
  int ox = (int)(tid % (2 * W));
  size_t t2 = tid / (2 * W);
  int oy = (int)(t2 % (2 * H));
  size_t t3 = t2 / (2 * H);
  int o = (int)(t3 % O);
  int b = (int)(t3 / O);
  int i = oy & 1, h = oy >> 1;
  int j = ox & 1, w2 = ox >> 1;
  out[tid] = res[(((size_t)b * H + h) * W + w2) * (size_t)(O * 4) + o * 4 + i * 2 + j];
}

// Plain 2D transpose: in[R, C] -> out[C, R]
__global__ void k_t2d(const float* __restrict__ in, float* __restrict__ out, int R, int Cc)
{
  size_t tid = (size_t)blockIdx.x * blockDim.x + threadIdx.x;
  if (tid >= (size_t)R * Cc) return;
  int r = (int)(tid / Cc);
  int c = (int)(tid % Cc);
  out[(size_t)c * R + r] = in[tid];
}

// BN folding: scale = g/sqrt(v+eps); shift = convb*scale + b - m*scale
__global__ void k_bn_prep(const float* __restrict__ g, const float* __restrict__ bb,
                          const float* __restrict__ m, const float* __restrict__ v,
                          const float* __restrict__ convb,
                          float* __restrict__ scale, float* __restrict__ shift, int n)
{
  int i = blockIdx.x * blockDim.x + threadIdx.x;
  if (i >= n) return;
  float inv = g[i] * rsqrtf(v[i] + 1e-5f);
  float cb = convb ? convb[i] : 0.f;
  scale[i] = inv;
  shift[i] = cb * inv + bb[i] - m[i] * inv;
}

// ---------------------------------------------------------------------------
// Host-side orchestration
// ---------------------------------------------------------------------------
struct MambaP {
  const float *A_log, *Dp, *conv_b, *conv_w, *dt_b, *dt_w, *in_proj, *out_proj, *x_proj;
};

// Assumed flat-input order: JAX pytree flatten (dict keys sorted).
// setup_inputs: feat0..feat3, params{...}, th, tw
// params sorted: bn1{b,g,m,v}, bn2{b,g,m,v}, cross(mamba), ct1_b, ct1_w,
//   ct2_b, ct2_w, final(mamba), level[0..3](mamba), proj_b, proj_bn{b,g,m,v},
//   proj_w, seg_b, seg_w
// mamba sorted: A_log, Dp, conv_b, conv_w, dt_b, dt_w, in_proj, out_proj, x_proj
enum {
  IN_FEAT0 = 0,
  IN_BN1_B = 4,  IN_BN1_G = 5,  IN_BN1_M = 6,  IN_BN1_V = 7,
  IN_BN2_B = 8,  IN_BN2_G = 9,  IN_BN2_M = 10, IN_BN2_V = 11,
  IN_CROSS = 12,
  IN_CT1_B = 21, IN_CT1_W = 22, IN_CT2_B = 23, IN_CT2_W = 24,
  IN_FINAL = 25,
  IN_LEVEL0 = 34,   // 4 x 9
  IN_PROJ_B = 70,
  IN_PROJBN_B = 71, IN_PROJBN_G = 72, IN_PROJBN_M = 73, IN_PROJBN_V = 74,
  IN_PROJ_W = 75,
  IN_SEG_B = 76, IN_SEG_W = 77
};

static MambaP load_mamba(void* const* d_in, int base) {
  MambaP p;
  p.A_log    = (const float*)d_in[base + 0];
  p.Dp       = (const float*)d_in[base + 1];
  p.conv_b   = (const float*)d_in[base + 2];
  p.conv_w   = (const float*)d_in[base + 3];
  p.dt_b     = (const float*)d_in[base + 4];
  p.dt_w     = (const float*)d_in[base + 5];
  p.in_proj  = (const float*)d_in[base + 6];
  p.out_proj = (const float*)d_in[base + 7];
  p.x_proj   = (const float*)d_in[base + 8];
  return p;
}

static inline void gemm(hipStream_t s, const float* A, int lda,
                        const float* Wt, int ldw, const float* bias,
                        const float* scale, const float* shift, int bnShift,
                        float* C, int ldc, int M, int N, int K, int epi) {
  dim3 grid((M + 15) / 16, (N + 127) / 128);   // 4 waves x 2 tiles x 16 cols
  k_gemm<<<grid, dim3(128), 0, s>>>(A, lda, Wt, ldw, bias, scale, shift, bnShift,
                                    C, ldc, M, N, K, epi);
}

static inline dim3 ew_grid(size_t n) { return dim3((unsigned)((n + 255) / 256)); }

static void run_mamba(hipStream_t s, const float* x_bcl, float* out_bcl,
                      int Bb, int D, int L, const MambaP& p,
                      float* seq, float* xz, float* xc, float* dbc,
                      float* dtb, float* yg, float* outseq)
{
  int di = 2 * D;
  int dr = (D + 15) / 16; if (dr < 1) dr = 1;
  int M = Bb * L;
  size_t nBCL = (size_t)Bb * D * L;
  size_t nXC  = (size_t)M * di;

  k_t_bcl_blc<<<ew_grid(nBCL), 256, 0, s>>>(x_bcl, seq, Bb, D, L);
  // xz = seq @ in_proj.T                 [M, 2di]
  gemm(s, seq, D, p.in_proj, D, nullptr, nullptr, nullptr, 0,
       xz, 2 * di, M, 2 * di, D, 0);
  // xc = silu(depthwise_conv(xh) + b)    [M, di]
  k_dwconv_silu<<<ew_grid(nXC), 256, 0, s>>>(xz, 2 * di, p.conv_w, p.conv_b,
                                             xc, Bb, L, di);
  // dbc = xc @ x_proj.T                  [M, dr+32]
  gemm(s, xc, di, p.x_proj, di, nullptr, nullptr, nullptr, 0,
       dbc, dr + 32, M, dr + 32, di, 0);
  // dt = softplus(dbc[:, :dr] @ dt_w.T + dt_b)   [M, di]
  gemm(s, dbc, dr + 32, p.dt_w, dr, p.dt_b, nullptr, nullptr, 0,
       dtb, di, M, di, dr, 1);
  // gated selective scan -> yg           [M, di]
  int nscan = Bb * di;
  k_scan<<<(nscan + 127) / 128, 128, 0, s>>>(dtb, xc, dbc, dr + 32, dr, dr + 16,
                                             xz, 2 * di, di, p.A_log, p.Dp,
                                             yg, Bb, L, di);
  // out_seq = yg @ out_proj.T            [M, D]
  gemm(s, yg, di, p.out_proj, di, nullptr, nullptr, nullptr, 0,
       outseq, D, M, D, di, 0);
  // out_bcl = residual + transpose(out_seq)
  k_t_blc_bcl_add<<<ew_grid(nBCL), 256, 0, s>>>(outseq, x_bcl, out_bcl, Bb, D, L);
}

extern "C" void kernel_launch(void* const* d_in, const int* in_sizes, int n_in,
                              void* d_out, int out_size, void* d_ws, size_t ws_size,
                              hipStream_t stream)
{
  (void)in_sizes; (void)n_in; (void)out_size; (void)ws_size;
  hipStream_t s = stream;
  const float* feat[4] = {(const float*)d_in[0], (const float*)d_in[1],
                          (const float*)d_in[2], (const float*)d_in[3]};
  const int Hs[4] = {48, 24, 12, 6};

  // ---- workspace carve (floats; ~197 MB total) ----
  float* wsp = (float*)d_ws;
  size_t off = 0;
  auto alloc = [&](size_t nf) { float* r = wsp + off; off += nf; return r; };
  float* seq    = alloc(2359296);   // max [B*L, D] staging
  float* xz     = alloc(18874368);  // cross: [4608, 2048]
  float* xc     = alloc(4718592);
  float* dbc    = alloc(2506752);
  float* dtb    = alloc(4718592);
  float* yg     = alloc(4718592);
  float* outseq = alloc(2359296);
  float* xcat   = alloc(2359296);   // (2, 512, 48, 48)
  float* y2     = alloc(2359296);   // (2, 32, 192, 192)
  float* r9a    = alloc(2359296);
  float* r9b    = alloc(1179648);
  float* fbuf   = alloc(589824);    // per-level mamba output (2,128,48,48 max)
  float* wT1    = alloc(32768);     // ct1 weights transposed (256,128)
  float* wT2    = alloc(8192);      // ct2 weights transposed (128,64)
  float* bnS0 = alloc(128); float* bnH0 = alloc(128);
  float* bnS1 = alloc(64);  float* bnH1 = alloc(64);
  float* bnS2 = alloc(32);  float* bnH2 = alloc(32);

  // ---- per-level mamba + resize into concat buffer ----
  for (int i = 0; i < 4; ++i) {
    MambaP p = load_mamba(d_in, IN_LEVEL0 + 9 * i);
    int H = Hs[i], L = H * H;
    run_mamba(s, feat[i], fbuf, 2, 128, L, p, seq, xz, xc, dbc, dtb, yg, outseq);
    size_t nout = (size_t)2 * 128 * 48 * 48;
    k_resize<<<ew_grid(nout), 256, 0, s>>>(fbuf, 2, 128, H, H,
                                           xcat, 512, 128 * i, 48, 48);
  }

  // ---- cross mamba on (2, 512, 48, 48), residual in place ----
  {
    MambaP p = load_mamba(d_in, IN_CROSS);
    run_mamba(s, xcat, xcat, 2, 512, 2304, p, seq, xz, xc, dbc, dtb, yg, outseq);
  }

  // ---- proj 1x1 conv (512->128) + BN + ReLU ----
  k_bn_prep<<<1, 128, 0, s>>>((const float*)d_in[IN_PROJBN_G], (const float*)d_in[IN_PROJBN_B],
                              (const float*)d_in[IN_PROJBN_M], (const float*)d_in[IN_PROJBN_V],
                              (const float*)d_in[IN_PROJ_B], bnS0, bnH0, 128);
  k_t_bcl_blc<<<ew_grid((size_t)2 * 512 * 2304), 256, 0, s>>>(xcat, seq, 2, 512, 2304);
  gemm(s, seq, 512, (const float*)d_in[IN_PROJ_W], 512, nullptr, bnS0, bnH0, 0,
       r9a, 128, 4608, 128, 512, 2);            // r9a = p1 [4608, 128]

  // ---- convT1 2x2 (128->64), 48->96, BN+ReLU fused in GEMM ----
  k_bn_prep<<<1, 64, 0, s>>>((const float*)d_in[IN_BN1_G], (const float*)d_in[IN_BN1_B],
                             (const float*)d_in[IN_BN1_M], (const float*)d_in[IN_BN1_V],
                             (const float*)d_in[IN_CT1_B], bnS1, bnH1, 64);
  k_t2d<<<ew_grid(128 * 256), 256, 0, s>>>((const float*)d_in[IN_CT1_W], wT1, 128, 256);
  gemm(s, r9a, 128, wT1, 128, nullptr, bnS1, bnH1, 2,
       r9b, 256, 4608, 256, 128, 2);            // r9b [4608, 256]
  k_scatter_convT<<<ew_grid((size_t)2 * 64 * 96 * 96), 256, 0, s>>>(r9b, r9a, 2, 64, 48, 48);

  // ---- convT2 2x2 (64->32), 96->192, BN+ReLU fused ----
  k_bn_prep<<<1, 32, 0, s>>>((const float*)d_in[IN_BN2_G], (const float*)d_in[IN_BN2_B],
                             (const float*)d_in[IN_BN2_M], (const float*)d_in[IN_BN2_V],
                             (const float*)d_in[IN_CT2_B], bnS2, bnH2, 32);
  k_t2d<<<ew_grid(64 * 128), 256, 0, s>>>((const float*)d_in[IN_CT2_W], wT2, 64, 128);
  k_t_bcl_blc<<<ew_grid((size_t)2 * 64 * 9216), 256, 0, s>>>(r9a, r9b, 2, 64, 9216);
  gemm(s, r9b, 64, wT2, 64, nullptr, bnS2, bnH2, 2,
       r9a, 128, 18432, 128, 64, 2);            // r9a [18432, 128]
  k_scatter_convT<<<ew_grid((size_t)2 * 32 * 192 * 192), 256, 0, s>>>(r9a, y2, 2, 32, 96, 96);

  // ---- final mamba on (2, 32, 192, 192), residual in place ----
  {
    MambaP p = load_mamba(d_in, IN_FINAL);
    run_mamba(s, y2, y2, 2, 32, 36864, p, seq, xz, xc, dbc, dtb, yg, outseq);
  }

  // ---- seg 1x1 conv (32->19); output already 192x192 == target ----
  k_t_bcl_blc<<<ew_grid((size_t)2 * 32 * 36864), 256, 0, s>>>(y2, seq, 2, 32, 36864);
  gemm(s, seq, 32, (const float*)d_in[IN_SEG_W], 32, (const float*)d_in[IN_SEG_B],
       nullptr, nullptr, 0, r9a, 19, 73728, 19, 32, 0);
  k_t_blc_bcl_add<<<ew_grid((size_t)2 * 19 * 36864), 256, 0, s>>>(
      r9a, nullptr, (float*)d_out, 2, 19, 36864);
}
